// GATClassifier_26431228740368
// MI455X (gfx1250) — matrix operand order repaired and checked by reading, not compiled, not run on hardware
//
#include <hip/hip_runtime.h>
#include <hip/hip_bf16.h>
#include <stdint.h>

#define TPB 256
#define NEG_SLOPE 0.2f
#define MT_PER_WAVE 4

typedef __attribute__((ext_vector_type(16))) _Float16 v16h;
typedef __attribute__((ext_vector_type(8)))  _Float16 v8h;
typedef __attribute__((ext_vector_type(8)))  float    v8f;

// ---------------------------------------------------------------- utilities
__global__ void k_fill_u32(uint32_t* __restrict__ p, long n, uint32_t v) {
  long i = (long)blockIdx.x * blockDim.x + threadIdx.x;
  if (i < n) p[i] = v;
}

__global__ void k_f32_to_f16(const float* __restrict__ in, _Float16* __restrict__ out, long n) {
  long i = (long)blockIdx.x * blockDim.x + threadIdx.x;
  if (i < n) out[i] = (_Float16)in[i];
}

// W[K][Nc] row-major f32  ->  Wt[NP][K] f16 (B in col-major, rows >= Nc zero-padded)
__global__ void k_w_transpose_f16(const float* __restrict__ W, _Float16* __restrict__ Wt,
                                  int K, int Nc, int NP) {
  int i = blockIdx.x * blockDim.x + threadIdx.x;
  if (i >= NP * K) return;
  int n = i / K, k = i - n * K;
  Wt[i] = (n < Nc) ? (_Float16)W[(long)k * Nc + n] : (_Float16)0.0f;
}

// ---------------------------------------------------------------- WMMA GEMM
// C[M][Np] = A[M][K](f16 row-major) * Bt[Np][K]^T (f16, B col-major) + bias
// One wave: 1 N-tile x MT_PER_WAVE M-tiles; B fragments preloaded in VGPRs and
// reused across the M-tiles (halves load bytes per v_wmma). K is a template
// parameter so bfrag[] indexing is static (no scratch spills).
template <int K>
__global__ void k_gemm_wmma(const _Float16* __restrict__ A,
                            const _Float16* __restrict__ Bt,
                            const float* __restrict__ bias, int nbias,
                            float* __restrict__ Cf, _Float16* __restrict__ Ch,
                            int M, int Np) {
  constexpr int NK = K / 32;             // 16x16x32 steps over K
  int wid  = (int)((blockIdx.x * blockDim.x + threadIdx.x) >> 5);
  int lane = threadIdx.x & 31;
  int ntiles = Np >> 4;
  int mtiles = M >> 4;
  int msuper = (mtiles + MT_PER_WAVE - 1) / MT_PER_WAVE;
  if (wid >= msuper * ntiles) return;    // wave-uniform: EXEC all-1s for WMMA
  int ms = wid / ntiles, nt = wid - ms * ntiles;
  int half = lane >> 4;
  int l16  = lane & 15;
  int col  = nt * 16 + l16;

  // preload B fragments for the whole K extent (lane group covers contiguous
  // K = k0+16*half .. +15 of column `col`)
  const _Float16* __restrict__ bcol = Bt + (long)col * K;
  v16h bfrag[NK];
#pragma unroll
  for (int k = 0; k < NK; ++k)
    bfrag[k] = *(const v16h*)(bcol + (k << 5) + 16 * half);

  float bv = 0.0f;
  if (bias) bv = (col < nbias) ? bias[col] : 0.0f;

#pragma unroll
  for (int i = 0; i < MT_PER_WAVE; ++i) {
    int mt = ms * MT_PER_WAVE + i;
    if (mt >= mtiles) break;             // wave-uniform guard (M remainder)
    int row = mt * 16 + l16;
    const _Float16* __restrict__ arow = A + (long)row * K;
    v8f acc = {};
#pragma unroll
    for (int k = 0; k < NK; ++k) {
      int k0 = k << 5;
      // A 16x32 fragment: lane half h holds contiguous K = k0+16h..+15
      v8h alo = *(const v8h*)(arow + k0 + 16 * half);
      v8h ahi = *(const v8h*)(arow + k0 + 16 * half + 8);
      v16h a = __builtin_shufflevector(alo, ahi,
                                       0,1,2,3,4,5,6,7,8,9,10,11,12,13,14,15);
      acc = __builtin_amdgcn_wmma_f32_16x16x32_f16(false, a, false, bfrag[k],
                                                   (short)0, acc, false, false);
    }
#pragma unroll
    for (int r = 0; r < 8; ++r) {
      int orow = mt * 16 + r + 8 * half; // D: VGPR r -> M = r + 8*half
      float v = acc[r] + bv;
      long idx = (long)orow * Np + col;
      if (Cf) Cf[idx] = v;
      if (Ch) Ch[idx] = (_Float16)v;
    }
  }
}

// ------------------------------------------------------- per-node attn logits
__global__ void k_logits(const float* __restrict__ h, const float* __restrict__ asrc,
                         const float* __restrict__ adst, float* __restrict__ ls,
                         float* __restrict__ ld, int N, int F) {
  int wid  = (int)((blockIdx.x * blockDim.x + threadIdx.x) >> 5);
  int lane = threadIdx.x & 31;
  if (wid >= N) return;
  int c = F >> 5;                        // features per lane (4 or 2)
  const float* hr = h + (long)wid * F + lane * c;
  float s0 = 0.f, s1 = 0.f;
  for (int i = 0; i < c; ++i) {
    float v = hr[i];
    s0 += v * asrc[lane * c + i];
    s1 += v * adst[lane * c + i];
  }
  for (int o = 16; o > 0; o >>= 1) {
    s0 += __shfl_down(s0, o, 32);
    s1 += __shfl_down(s1, o, 32);
  }
  if (lane == 0) { ls[wid] = s0; ld[wid] = s1; }
}

// ------------------------------------------------------------- edge kernels
__device__ __forceinline__ void edge_sd(const int* __restrict__ ei, int E, int idx,
                                        int& s, int& d) {
  if (idx < E) { s = ei[idx]; d = ei[E + idx]; }
  else         { s = d = idx - E; }      // self-loops appended
}
__device__ __forceinline__ float leaky(float e) { return e > 0.f ? e : NEG_SLOPE * e; }
__device__ __forceinline__ int   f2mono(float f) { int b = __float_as_int(f); return b >= 0 ? b : (b ^ 0x7fffffff); }
__device__ __forceinline__ float mono2f(int m)   { return __int_as_float(m >= 0 ? m : (m ^ 0x7fffffff)); }

__global__ void k_edge_max(const int* __restrict__ ei, int E, int ET,
                           const float* __restrict__ ls, const float* __restrict__ ld,
                           int* __restrict__ mkey) {
  int idx = blockIdx.x * blockDim.x + threadIdx.x;
  if (idx >= ET) return;
  int s, d; edge_sd(ei, E, idx, s, d);
  atomicMax(&mkey[d], f2mono(leaky(ls[s] + ld[d])));
}

__global__ void k_edge_sum(const int* __restrict__ ei, int E, int ET,
                           const float* __restrict__ ls, const float* __restrict__ ld,
                           const int* __restrict__ mkey, float* __restrict__ den) {
  int idx = blockIdx.x * blockDim.x + threadIdx.x;
  if (idx >= ET) return;
  int s, d; edge_sd(ei, E, idx, s, d);
  float e = leaky(ls[s] + ld[d]);
  atomicAdd(&den[d], expf(e - mono2f(mkey[d])));
}

__global__ void k_edge_alpha(const int* __restrict__ ei, int E, int ET,
                             const float* __restrict__ ls, const float* __restrict__ ld,
                             const int* __restrict__ mkey, const float* __restrict__ den,
                             float* __restrict__ alpha) {
  int idx = blockIdx.x * blockDim.x + threadIdx.x;
  if (idx >= ET) return;
  int s, d; edge_sd(ei, E, idx, s, d);
  float e = leaky(ls[s] + ld[d]);
  alpha[idx] = expf(e - mono2f(mkey[d])) / den[d];
}

// one lane per (edge, 4 features); lshift = log2(F/4) lanes per edge
__global__ void k_edge_scatter(const int* __restrict__ ei, int E, int ET,
                               const float* __restrict__ alpha,
                               const float* __restrict__ h, float* __restrict__ agg,
                               int F, int lshift) {
  long t = (long)blockIdx.x * blockDim.x + threadIdx.x;
  long e = t >> lshift;
  if (e >= ET) return;
  int fo = (int)(t & ((1 << lshift) - 1)) << 2;
  int s, d; edge_sd(ei, E, (int)e, s, d);
  float al = alpha[e];                   // broadcast load, uniform across the edge's 2^lshift lanes
  const float4 hv = *(const float4*)(h + (long)s * F + fo);
  float* ag = agg + (long)d * F + fo;
  atomicAdd(ag + 0, hv.x * al);
  atomicAdd(ag + 1, hv.y * al);
  atomicAdd(ag + 2, hv.z * al);
  atomicAdd(ag + 3, hv.w * al);
}

// z = relu(agg + b) -> f16 (feeds next WMMA GEMM)
__global__ void k_finalize(const float* __restrict__ agg, const float* __restrict__ bias,
                           _Float16* __restrict__ zh, long n, int F) {
  long i = (long)blockIdx.x * blockDim.x + threadIdx.x;
  if (i >= n) return;
  float v = agg[i] + bias[(int)(i % F)];
  zh[i] = (_Float16)(v > 0.f ? v : 0.f);
}

__global__ void k_logsoftmax(const float* __restrict__ pre, float* __restrict__ out, int N) {
  int n = blockIdx.x * blockDim.x + threadIdx.x;
  if (n >= N) return;
  const float* r = pre + (long)n * 16;   // padded row stride 16, first 10 valid
  float mx = r[0];
  for (int i = 1; i < 10; ++i) mx = fmaxf(mx, r[i]);
  float s = 0.f;
  for (int i = 0; i < 10; ++i) s += expf(r[i] - mx);
  float l = logf(s);
  for (int i = 0; i < 10; ++i) out[(long)n * 10 + i] = r[i] - mx - l;
}

// ---------------------------------------------------------------- launcher
static inline int cdiv(long a, long b) { return (int)((a + b - 1) / b); }

extern "C" void kernel_launch(void* const* d_in, const int* in_sizes, int n_in,
                              void* d_out, int out_size, void* d_ws, size_t ws_size,
                              hipStream_t stream) {
  (void)n_in; (void)out_size; (void)ws_size;
  const float* x   = (const float*)d_in[0];
  const int*   ei  = (const int*)  d_in[1];
  const float* W1  = (const float*)d_in[2];
  const float* as1 = (const float*)d_in[3];
  const float* ad1 = (const float*)d_in[4];
  const float* b1  = (const float*)d_in[5];
  const float* Wl  = (const float*)d_in[6];
  const float* bl  = (const float*)d_in[7];
  const float* W2  = (const float*)d_in[8];
  const float* as2 = (const float*)d_in[9];
  const float* ad2 = (const float*)d_in[10];
  const float* b2  = (const float*)d_in[11];
  const float* Wm  = (const float*)d_in[12];
  const float* bm  = (const float*)d_in[13];

  const int F1 = 128, F2 = 64;
  const int N  = in_sizes[0] / F1;       // 50000 (multiple of 16)
  const int E  = in_sizes[1] / 2;        // 800000
  const int ET = E + N;                  // + self loops

  char* ws = (char*)d_ws;
  size_t off = 0;
  auto alloc = [&](size_t bytes) -> void* {
    void* p = ws + off;
    off = (off + bytes + 255) & ~(size_t)255;
    return p;
  };

  // region A: xh -> z1h -> pre    (N*F1*2 bytes >= N*16*4)
  _Float16* xh  = (_Float16*)alloc((size_t)N * F1 * 2);
  // region B: h1 -> {hmid_h, h2}  (N*F1*4 bytes)
  float*    h1  = (float*)alloc((size_t)N * F1 * 4);
  // region G: agg1 -> {agg2, z2h} (N*F1*4 bytes)
  float*    agg1 = (float*)alloc((size_t)N * F1 * 4);
  float* ls1 = (float*)alloc((size_t)N * 4);
  float* ld1 = (float*)alloc((size_t)N * 4);
  int*   m1  = (int*)  alloc((size_t)N * 4);
  float* dn1 = (float*)alloc((size_t)N * 4);
  float* ls2 = (float*)alloc((size_t)N * 4);
  float* ld2 = (float*)alloc((size_t)N * 4);
  int*   m2  = (int*)  alloc((size_t)N * 4);
  float* dn2 = (float*)alloc((size_t)N * 4);
  float* al1 = (float*)alloc((size_t)ET * 4);
  float* al2 = (float*)alloc((size_t)ET * 4);
  _Float16* W1t = (_Float16*)alloc(128 * 128 * 2);
  _Float16* Wlt = (_Float16*)alloc(128 * 128 * 2);
  _Float16* W2t = (_Float16*)alloc(64 * 128 * 2);
  _Float16* Wmt = (_Float16*)alloc(16 * 64 * 2);
  // aliases (liveness-checked)
  _Float16* z1h   = (_Float16*)xh;
  _Float16* hmidh = (_Float16*)h1;
  float*    h2    = (float*)((char*)h1 + (size_t)N * F1 * 2);
  float*    agg2  = agg1;
  _Float16* z2h   = (_Float16*)((char*)agg1 + (size_t)N * F2 * 4);
  float*    pre   = (float*)xh;

  const uint32_t NEG_INF_KEY = 0x807FFFFFu; // f2mono(-inf)
  const int mtiles = N / 16;
  const int msuper = (mtiles + MT_PER_WAVE - 1) / MT_PER_WAVE;

  // ---- prep: f16 activations + transposed f16 weights
  k_f32_to_f16<<<cdiv((long)N * F1, TPB), TPB, 0, stream>>>(x, xh, (long)N * F1);
  k_w_transpose_f16<<<cdiv(128 * 128, TPB), TPB, 0, stream>>>(W1, W1t, 128, 128, 128);
  k_w_transpose_f16<<<cdiv(128 * 128, TPB), TPB, 0, stream>>>(Wl, Wlt, 128, 128, 128);
  k_w_transpose_f16<<<cdiv(64 * 128, TPB), TPB, 0, stream>>>(W2, W2t, 128, 64, 64);
  k_w_transpose_f16<<<cdiv(16 * 64, TPB), TPB, 0, stream>>>(Wm, Wmt, 64, 10, 16);

  // ---- layer 1: h1 = x @ W1
  {
    long waves = (long)msuper * (F1 / 16);
    k_gemm_wmma<128><<<cdiv(waves * 32, TPB), TPB, 0, stream>>>(
        xh, W1t, nullptr, 0, h1, nullptr, N, F1);
  }
  k_logits<<<cdiv((long)N * 32, TPB), TPB, 0, stream>>>(h1, as1, ad1, ls1, ld1, N, F1);
  k_fill_u32<<<cdiv(N, TPB), TPB, 0, stream>>>((uint32_t*)m1, N, NEG_INF_KEY);
  k_fill_u32<<<cdiv(N, TPB), TPB, 0, stream>>>((uint32_t*)dn1, N, 0u);
  k_fill_u32<<<cdiv((long)N * F1, TPB), TPB, 0, stream>>>((uint32_t*)agg1, (long)N * F1, 0u);
  k_edge_max  <<<cdiv(ET, TPB), TPB, 0, stream>>>(ei, E, ET, ls1, ld1, m1);
  k_edge_sum  <<<cdiv(ET, TPB), TPB, 0, stream>>>(ei, E, ET, ls1, ld1, m1, dn1);
  k_edge_alpha<<<cdiv(ET, TPB), TPB, 0, stream>>>(ei, E, ET, ls1, ld1, m1, dn1, al1);
  k_edge_scatter<<<cdiv((long)ET * 32, TPB), TPB, 0, stream>>>(ei, E, ET, al1, h1, agg1, F1, 5);
  k_finalize<<<cdiv((long)N * F1, TPB), TPB, 0, stream>>>(agg1, b1, z1h, (long)N * F1, F1);

  // ---- hidden linear: hmid = z1 @ Wl + bl (f16 out feeds next GEMM)
  {
    long waves = (long)msuper * (F1 / 16);
    k_gemm_wmma<128><<<cdiv(waves * 32, TPB), TPB, 0, stream>>>(
        z1h, Wlt, bl, F1, nullptr, hmidh, N, F1);
  }

  // ---- layer 2: h2 = hmid @ W2
  {
    long waves = (long)msuper * (F2 / 16);
    k_gemm_wmma<128><<<cdiv(waves * 32, TPB), TPB, 0, stream>>>(
        hmidh, W2t, nullptr, 0, h2, nullptr, N, F2);
  }
  k_logits<<<cdiv((long)N * 32, TPB), TPB, 0, stream>>>(h2, as2, ad2, ls2, ld2, N, F2);
  k_fill_u32<<<cdiv(N, TPB), TPB, 0, stream>>>((uint32_t*)m2, N, NEG_INF_KEY);
  k_fill_u32<<<cdiv(N, TPB), TPB, 0, stream>>>((uint32_t*)dn2, N, 0u);
  k_fill_u32<<<cdiv((long)N * F2, TPB), TPB, 0, stream>>>((uint32_t*)agg2, (long)N * F2, 0u);
  k_edge_max  <<<cdiv(ET, TPB), TPB, 0, stream>>>(ei, E, ET, ls2, ld2, m2);
  k_edge_sum  <<<cdiv(ET, TPB), TPB, 0, stream>>>(ei, E, ET, ls2, ld2, m2, dn2);
  k_edge_alpha<<<cdiv(ET, TPB), TPB, 0, stream>>>(ei, E, ET, ls2, ld2, m2, dn2, al2);
  k_edge_scatter<<<cdiv((long)ET * 16, TPB), TPB, 0, stream>>>(ei, E, ET, al2, h2, agg2, F2, 4);
  k_finalize<<<cdiv((long)N * F2, TPB), TPB, 0, stream>>>(agg2, b2, z2h, (long)N * F2, F2);

  // ---- classifier head: pre = z2 @ Wm + bm (N padded to 16 cols, bias guarded)
  {
    long waves = (long)msuper * 1;
    k_gemm_wmma<64><<<cdiv(waves * 32, TPB), TPB, 0, stream>>>(
        z2h, Wmt, bm, 10, pre, nullptr, N, 16);
  }
  k_logsoftmax<<<cdiv(N, TPB), TPB, 0, stream>>>(pre, (float*)d_out, N);
}
